// LossFunction_33285996544604
// MI455X (gfx1250) — matrix operand order repaired and checked by reading, not compiled, not run on hardware
//
#include <hip/hip_runtime.h>
#include <stdint.h>

// ---------------- problem constants (from setup_inputs) ----------------
#define NA 2048
#define NZ 1024
#define NN (NA*NZ)          // 2097152
#define PTS 65536
#define HID 128
#define NSTEPS 7            // Nt=8 -> 7 scan steps (Nt is fixed by setup_inputs)

#define DT_     0.01f
#define ALPHA_  0.36f
#define DELTA_  0.05f
#define THETA_  0.25f
#define HATZ_   1.0f
#define SIGZ_   0.1f
#define ETA_    0.25f
#define HATA_   1.0f
#define SIG_    0.05f
#define RHO_    0.05f
#define ZLO_    0.5f
#define ZHI_    1.5f
#define ALO_    0.5f
#define AHI_    1.5f

typedef __attribute__((ext_vector_type(16))) _Float16 v16h;
typedef __attribute__((ext_vector_type(8)))  _Float16 v8h;
typedef __attribute__((ext_vector_type(8)))  float    v8f;
typedef __attribute__((ext_vector_type(4)))  unsigned int v4u;
typedef __attribute__((ext_vector_type(8)))  int      v8i;
typedef __attribute__((ext_vector_type(4)))  int      v4i;

// ---------------- deterministic counter-based RNG ----------------
__device__ __forceinline__ uint32_t mix_u32(uint64_t x) {
    x ^= x >> 33; x *= 0xff51afd7ed558ccdULL;
    x ^= x >> 33; x *= 0xc4ceb9fe1a85ec53ULL;
    x ^= x >> 33;
    return (uint32_t)x;
}

__device__ __forceinline__ void rng_normals4(int ti, int e, float n[4]) {
    uint64_t base = (((uint64_t)(ti + 1)) << 40) ^ ((uint64_t)(uint32_t)e * 0x9E3779B97F4A7C15ULL);
    uint32_t u0 = mix_u32(base + 0x0123456789ABCDEFULL);
    uint32_t u1 = mix_u32(base ^ 0xDEADBEEFCAFEBABEULL);
    uint32_t u2 = mix_u32(base + 0x5555555555555555ULL);
    uint32_t u3 = mix_u32(base ^ 0x0F0F0F0F0F0F0F0FULL);
    const float inv32 = 2.3283064365386963e-10f; // 2^-32
    float f0 = ((float)u0 + 0.5f) * inv32;
    float f1 = ((float)u1 + 0.5f) * inv32;
    float f2 = ((float)u2 + 0.5f) * inv32;
    float f3 = ((float)u3 + 0.5f) * inv32;
    float r0 = sqrtf(-2.0f * logf(f0));
    float r1 = sqrtf(-2.0f * logf(f2));
    float s0, c0, s1, c1;
    __sincosf(6.28318530718f * f1, &s0, &c0);
    __sincosf(6.28318530718f * f3, &s1, &c1);
    n[0] = r0 * c0; n[1] = r0 * s0;
    n[2] = r1 * c1; n[3] = r1 * s1;
}

// ---------------- reductions (deterministic two-stage) ----------------
__global__ __launch_bounds__(256) void k_reduce_partial(const float* __restrict__ x,
                                                        float* __restrict__ partials) {
    __shared__ float s[256];
    int base = blockIdx.x * 1024;
    float v = 0.f;
    for (int i = threadIdx.x; i < 1024; i += 256) v += x[base + i];
    s[threadIdx.x] = v;
    __syncthreads();
    for (int w = 128; w > 0; w >>= 1) {
        if (threadIdx.x < w) s[threadIdx.x] += s[threadIdx.x + w];
        __syncthreads();
    }
    if (threadIdx.x == 0) partials[blockIdx.x] = s[0];
}

// mode 0: store K = mean.  mode 1: L = mean; store c1 = a*C/K, c2 = (1-a)*C/L.
__global__ __launch_bounds__(256) void k_reduce_final(const float* __restrict__ partials,
                                                      float* __restrict__ scal, int mode) {
    __shared__ float s[256];
    float v = 0.f;
    for (int i = threadIdx.x; i < 2048; i += 256) v += partials[i];
    s[threadIdx.x] = v;
    __syncthreads();
    for (int w = 128; w > 0; w >>= 1) {
        if (threadIdx.x < w) s[threadIdx.x] += s[threadIdx.x + w];
        __syncthreads();
    }
    if (threadIdx.x == 0) {
        float mean = s[0] * (1.0f / (float)NN);
        if (mode == 0) {
            scal[0] = mean;                       // K
        } else {
            float K = scal[0], L = mean;
            float C = powf(K, ALPHA_) * powf(L, 1.0f - ALPHA_);
            scal[1] = ALPHA_ * C / K;             // r = A*c1 - delta
            scal[2] = (1.0f - ALPHA_) * C / L;    // w = A*c2
        }
    }
}

// ---------------- W2 -> f16 transposed (done once) ----------------
__global__ __launch_bounds__(256) void k_w2_transpose(const float* __restrict__ W2,
                                                      _Float16* __restrict__ w2t) {
    int g = blockIdx.x * 256 + threadIdx.x;     // 16384 total
    if (g < HID * HID) {
        int j = g >> 7, k = g & 127;            // w2t[j][k] = W2[k][j]
        w2t[g] = (_Float16)W2[k * HID + j];
    }
}

// ---------------- step pass 1: OU + central diffs + Euler-Maruyama ----------------
__global__ __launch_bounds__(256) void k_step1(const float* __restrict__ a,
                                               float* __restrict__ z,
                                               float* __restrict__ A,
                                               const float* __restrict__ xi,
                                               float* __restrict__ xi1,
                                               float* __restrict__ lam,
                                               const float* __restrict__ scal,
                                               int ti) {
    int e = blockIdx.x * 256 + threadIdx.x;     // exactly NN threads
    int i = e >> 10;           // asset index (axis 0), Nz = 1024
    int j = e & (NZ - 1);      // z index (axis 1)

    float c1 = scal[1], c2 = scal[2];
    float zv = z[e], Av = A[e], xiv = xi[e], av = a[e];

    float n[4];
    rng_normals4(ti, e, n);
    float tif = (float)ti;
    float dWj = n[0] * (tif + 1.0f) * DT_ - n[1] * tif * DT_;
    float dWk = n[2] * (tif + 1.0f) * DT_ - n[3] * tif * DT_;

    float r = Av * c1 - DELTA_;
    float w = Av * c2;
    float Lam = r * av + w * zv - xiv;

    float zn = zv + THETA_ * (HATZ_ - zv) * DT_ + SIGZ_ * dWj;
    zn = fminf(fmaxf(zn, ZLO_), ZHI_);
    float An = Av + ETA_ * (HATA_ - Av) * DT_ + SIG_ * dWk;
    An = fminf(fmaxf(An, ALO_), AHI_);

    // central differences with wraparound (jnp.roll semantics)
    int jp = (j + 1) & (NZ - 1), jm = (j + NZ - 1) & (NZ - 1);
    int ip = (i + 1) & (NA - 1), im = (i + NA - 1) & (NA - 1);
    float xi_z = 0.5f * (xi[i * NZ + jp] - xi[i * NZ + jm]);
    float xi_A = 0.5f * (xi[ip * NZ + j] - xi[im * NZ + j]);

    float x1 = xiv + ((RHO_ - r) * xiv - Lam) * DT_
             + (-SIGZ_ * xi_z) * dWj + (-SIG_ * xi_A) * dWk;

    z[e]   = zn;   // z/A updated in place (L already reduced from old z)
    A[e]   = An;
    xi1[e] = x1;
    lam[e] = Lam;
}

// ---------------- step pass 2: upwind transport along asset axis ----------------
__global__ __launch_bounds__(256) void k_step2(const float* __restrict__ xi1,
                                               const float* __restrict__ lam,
                                               float* __restrict__ xi) {
    int e = blockIdx.x * 256 + threadIdx.x;
    int i = e >> 10, j = e & (NZ - 1);
    float x1 = xi1[e];
    float L  = lam[e];
    int ip = (i + 1) & (NA - 1), im = (i + NA - 1) & (NA - 1);
    float fwd = xi1[ip * NZ + j] - x1;
    float bwd = x1 - xi1[im * NZ + j];
    xi[e] = x1 + DT_ * L * ((L > 0.f) ? fwd : bwd);
}

// ---------------- MLP + loss: TDM staging + WMMA f16 hidden GEMM ----------------
// Block = 128 points, 256 threads (8 wave32). Each wave:
//  1) issues one tensor_load_to_lds moving its 4KB slice of W2^T into LDS (TDM,
//     overlapped with the gather + layer-1 VALU work),
//  2) owns a 16-row M-tile of the 128x128x128 GEMM: 8 N-tiles x 4 K-chunks
//     = 32 v_wmma_f32_16x16x32_f16 with accumulators held in VGPRs.
__global__ __launch_bounds__(256) void k_mlp_loss(const float* __restrict__ a,
                                                  const float* __restrict__ z,
                                                  const float* __restrict__ A,
                                                  const float* __restrict__ xi,
                                                  const float* __restrict__ W1,
                                                  const float* __restrict__ b1,
                                                  const _Float16* __restrict__ w2t,
                                                  const float* __restrict__ b2,
                                                  const float* __restrict__ W3,
                                                  const float* __restrict__ b3,
                                                  const long long* __restrict__ index,
                                                  float* __restrict__ loss_partial) {
    __shared__ __attribute__((aligned(128))) _Float16 sH[128 * 128];   // h1 -> later h2
    __shared__ __attribute__((aligned(128))) _Float16 sB[128 * 128];   // W2^T (n-major)
    __shared__ float sW1[4 * 128];
    __shared__ float sb1[128], sb2[128], sW3[128];
    __shared__ float sgz[128], sga[128], sgA[128], sgxi[128];
    __shared__ float sErr[128];

    const int t = threadIdx.x;
    const int wave = t >> 5, lane = t & 31;

    // ---- TDM: each wave DMAs its 4KB (2048-half) slice of W2^T into sB ----
    {
        uint32_t wv = (uint32_t)__builtin_amdgcn_readfirstlane(wave);
        uint32_t lds_base = (uint32_t)(uintptr_t)(void*)&sB[0] + wv * 4096u;
        uint64_t gaddr = (uint64_t)(uintptr_t)w2t + (uint64_t)wv * 4096u;

        v4u g0;
        g0[0] = 1u;                                   // count=1, user-mode load
        g0[1] = lds_base;                             // lds_addr (bytes)
        g0[2] = (uint32_t)gaddr;                      // global_addr[31:0]
        g0[3] = (uint32_t)((gaddr >> 32) & 0x01FFFFFFu) | (2u << 30); // addr[56:32] | type=2

        v8i g1;
        g1[0] = 0x00010000;                           // data_size=2B, wg_mask=0
        g1[1] = (int)(2048u << 16);                   // tensor_dim0[15:0]=2048
        g1[2] = (int)(1u << 16);                      // tensor_dim1=1
        g1[3] = (int)(2048u << 16);                   // tile_dim0=2048
        g1[4] = 1;                                    // tile_dim1=1, tile_dim2=0
        g1[5] = 2048;                                 // tensor_dim0_stride (elems)
        g1[6] = 0;
        g1[7] = 0;

        v4i gz4 = {};                                 // <=2D tensor: groups 2/3 unused
        v8i gz8 = {};                                 // extra group (clang-23 6-arg form)
        __builtin_amdgcn_tensor_load_to_lds(g0, g1, gz4, gz4, gz8, 0);
    }

    // stage consts (overlaps with TDM)
    for (int i = t; i < 512; i += 256) sW1[i] = W1[i];
    if (t < 128) { sb1[t] = b1[t]; sb2[t] = b2[t]; sW3[t] = W3[t]; }

    // gather inputs: x.T.flat[idx] == x[idx % Na, idx / Na]
    if (t < 128) {
        int p = blockIdx.x * 128 + t;
        long long idx = index[2 * p] + index[2 * p + 1];
        int row = (int)(idx & (NA - 1));
        int col = (int)(idx >> 11);
        int off = row * NZ + col;
        sgz[t] = z[off]; sga[t] = a[off]; sgA[t] = A[off]; sgxi[t] = xi[off];
    }
    __syncthreads();

    // layer 1 (K=4, VALU): 2 threads per point, 64 cols each (TDM still in flight)
    {
        int p = t & 127, jh = t >> 7;
        float gz = sgz[p], ga = sga[p], gA = sgA[p];
        #pragma unroll 4
        for (int jj = 0; jj < 64; ++jj) {
            int jc = jh * 64 + jj;
            float h = gz * sW1[0 * 128 + jc]
                    + ga * (sW1[1 * 128 + jc] + sW1[2 * 128 + jc])
                    + gA * sW1[3 * 128 + jc] + sb1[jc];
            sH[p * 128 + jc] = (_Float16)tanhf(h);
        }
    }
    __builtin_amdgcn_s_wait_tensorcnt(0);   // my slice of W2^T has landed
    __syncthreads();                        // everyone's slice has landed

    // layer 2: 128x128x128 GEMM via v_wmma_f32_16x16x32_f16
    const int m0 = wave * 16;
    const int hsel = lane >> 4;      // lane-half select
    const int lmod = lane & 15;

    v8f acc[8] = {};
    #pragma unroll
    for (int kc = 0; kc < 4; ++kc) {
        const int kb = kc * 32;
        // A fragment: row = m0+lmod; halves: K in {kb+hsel*8..+7} and {kb+16+hsel*8..+7}
        const _Float16* arow = &sH[(m0 + lmod) * 128 + kb + hsel * 8];
        v8h lo = *(const v8h*)(arow);
        v8h hi = *(const v8h*)(arow + 16);
        v16h af = __builtin_shufflevector(lo, hi, 0,1,2,3,4,5,6,7,8,9,10,11,12,13,14,15);
        #pragma unroll
        for (int n = 0; n < 8; ++n) {
            // B fragment: col n*16+lmod; halves: K in {kb+hsel*8..+7} and {kb+16+hsel*8..+7}
            const int ncol = n * 16 + lmod;
            const _Float16* bcol = &sB[ncol * 128 + kb + hsel * 8];
            v8h blo = *(const v8h*)(bcol);
            v8h bhi = *(const v8h*)(bcol + 16);
            v16h bf = __builtin_shufflevector(blo, bhi, 0,1,2,3,4,5,6,7,8,9,10,11,12,13,14,15);
            acc[n] = __builtin_amdgcn_wmma_f32_16x16x32_f16(
                false, af, false, bf, (short)0, acc[n], false, false);
        }
    }
    __syncthreads();   // everyone done reading sH before overwrite

    // h2 = tanh(acc + b2) written back over sH (f16)
    #pragma unroll
    for (int n = 0; n < 8; ++n) {
        const int ncol = n * 16 + lmod;
        float bias = sb2[ncol];
        #pragma unroll
        for (int r = 0; r < 8; ++r) {
            int row = m0 + r + 8 * hsel;
            sH[row * 128 + ncol] = (_Float16)tanhf(acc[n][r] + bias);
        }
    }
    __syncthreads();

    // layer 3 (N=1 dot) + squared error
    if (t < 128) {
        float s = 0.f;
        #pragma unroll 4
        for (int k = 0; k < 128; ++k) s += (float)sH[t * 128 + k] * sW3[k];
        float pred = s + b3[0];
        float d = sgxi[t] - pred;
        sErr[t] = d * d;
    }
    __syncthreads();
    if (t == 0) {
        float s = 0.f;
        for (int i = 0; i < 128; ++i) s += sErr[i];
        loss_partial[blockIdx.x] = s;
    }
}

// ---------------- final loss: sum 7*512 partials, / (P * (Nt-1)) ----------------
__global__ __launch_bounds__(256) void k_final(const float* __restrict__ lossp,
                                               const int* __restrict__ NtP,
                                               float* __restrict__ out) {
    __shared__ float s[256];
    float v = 0.f;
    for (int i = threadIdx.x; i < NSTEPS * 512; i += 256) v += lossp[i];
    s[threadIdx.x] = v;
    __syncthreads();
    for (int w = 128; w > 0; w >>= 1) {
        if (threadIdx.x < w) s[threadIdx.x] += s[threadIdx.x + w];
        __syncthreads();
    }
    if (threadIdx.x == 0) {
        int steps = NtP[0] - 1;
        if (steps < 1 || steps > NSTEPS) steps = NSTEPS;
        out[0] = s[0] / ((float)PTS * (float)steps);
    }
}

// ---------------- host launcher ----------------
extern "C" void kernel_launch(void* const* d_in, const int* in_sizes, int n_in,
                              void* d_out, int out_size, void* d_ws, size_t ws_size,
                              hipStream_t stream) {
    const float* a   = (const float*)d_in[0];
    const float* z0  = (const float*)d_in[1];
    const float* A0  = (const float*)d_in[2];
    const float* xi0 = (const float*)d_in[3];
    const float* W1  = (const float*)d_in[4];
    const float* b1  = (const float*)d_in[5];
    const float* W2  = (const float*)d_in[6];
    const float* b2  = (const float*)d_in[7];
    const float* W3  = (const float*)d_in[8];
    const float* b3  = (const float*)d_in[9];
    const long long* index = (const long long*)d_in[10];
    const int* NtP = (const int*)d_in[11];

    // workspace layout (floats)
    float* ws   = (float*)d_ws;
    float* z    = ws;
    float* A    = ws + (size_t)NN;
    float* xi   = ws + 2 * (size_t)NN;
    float* xi1  = ws + 3 * (size_t)NN;
    float* lam  = ws + 4 * (size_t)NN;
    float* part = ws + 5 * (size_t)NN;           // 2048
    float* scal = part + 2048;                   // 16
    float* lossp = scal + 16;                    // 7*512
    _Float16* w2t = (_Float16*)(lossp + NSTEPS * 512);   // 16384 halfs

    size_t need = (5 * (size_t)NN + 2048 + 16 + NSTEPS * 512) * sizeof(float)
                + (size_t)HID * HID * sizeof(_Float16);
    if (ws_size < need) return;   // deterministic no-op if workspace too small

    // state init (graph-capture-safe d2d copies)
    (void)hipMemcpyAsync(z,  z0,  (size_t)NN * sizeof(float), hipMemcpyDeviceToDevice, stream);
    (void)hipMemcpyAsync(A,  A0,  (size_t)NN * sizeof(float), hipMemcpyDeviceToDevice, stream);
    (void)hipMemcpyAsync(xi, xi0, (size_t)NN * sizeof(float), hipMemcpyDeviceToDevice, stream);

    k_w2_transpose<<<64, 256, 0, stream>>>(W2, w2t);

    // K = mean(a), constant across steps
    k_reduce_partial<<<2048, 256, 0, stream>>>(a, part);
    k_reduce_final<<<1, 256, 0, stream>>>(part, scal, 0);

    for (int ti = 0; ti < NSTEPS; ++ti) {
        // L = mean(z) (old z), then price scalars c1,c2
        k_reduce_partial<<<2048, 256, 0, stream>>>(z, part);
        k_reduce_final<<<1, 256, 0, stream>>>(part, scal, 1);

        k_step1<<<NN / 256, 256, 0, stream>>>(a, z, A, xi, xi1, lam, scal, ti);
        k_step2<<<NN / 256, 256, 0, stream>>>(xi1, lam, xi);

        k_mlp_loss<<<PTS / 128, 256, 0, stream>>>(a, z, A, xi, W1, b1, w2t, b2,
                                                  W3, b3, index, lossp + ti * 512);
    }

    k_final<<<1, 256, 0, stream>>>(lossp, NtP, (float*)d_out);
}
